// KPN_11879879544360
// MI455X (gfx1250) — matrix operand order repaired and checked by reading, not compiled, not run on hardware
//
#include <hip/hip_runtime.h>

// ---------------------------------------------------------------------------
// CDNA5 / gfx1250 U-Net. NHWC bf16 activations, implicit-GEMM conv5x5 via
// v_wmma_f32_16x16x32_bf16, A-tile staged with global_load_async_to_lds_b64
// (double-buffered, ASYNCcnt-synchronized), fused BN+ReLU epilogue.
// ---------------------------------------------------------------------------

typedef __attribute__((ext_vector_type(16))) __bf16 v16bf;
typedef __attribute__((ext_vector_type(8)))  __bf16 v8bf;
typedef __attribute__((ext_vector_type(8)))  float  v8f;

#define BN_INV 0.9999950000374997f   // 1/sqrt(1+1e-5)

// ---------------------------------------------------------------------------
// zero fill (bf16)
// ---------------------------------------------------------------------------
__global__ void k_zero_bf16(__bf16* __restrict__ p, int n) {
    int i = blockIdx.x * blockDim.x + threadIdx.x;
    if (i < n) p[i] = (__bf16)0.0f;
}

// ---------------------------------------------------------------------------
// input: NCHW fp32 (2,3,512,512) -> NHWC bf16 (2,512,512,32), chans 3..31 = 0
// ---------------------------------------------------------------------------
__global__ void k_cvt_pad(const float* __restrict__ in, __bf16* __restrict__ out,
                          int n) {
    int idx = blockIdx.x * blockDim.x + threadIdx.x;
    if (idx >= n) return;
    const int HW = 512 * 512;
    int c   = idx & 31;
    int pix = idx >> 5;              // b*HW + p
    int b   = pix >> 18;             // / HW
    int p   = pix & (HW - 1);
    float v = (c < 3) ? in[(size_t)b * 3 * HW + (size_t)c * HW + p] : 0.0f;
    out[idx] = (__bf16)v;
}

// ---------------------------------------------------------------------------
// Pack weights (O,C,5,5) fp32 -> bf16 [K/32][Opad][32], K = t*Cr + c
// (tap-major, channel-minor: matches the NHWC A-tile K ordering).
// ---------------------------------------------------------------------------
__global__ void k_pack_w(const float* __restrict__ w, __bf16* __restrict__ wp,
                         int C, int Cr, int O, int Opad, int total) {
    int idx = blockIdx.x * blockDim.x + threadIdx.x;
    if (idx >= total) return;
    int kk = idx & 31;
    int o  = (idx >> 5) % Opad;
    int kt = idx / (32 * Opad);
    int k  = kt * 32 + kk;
    int t  = k / Cr;                 // tap 0..24
    int c  = k - t * Cr;             // channel
    float v = 0.0f;
    if (c < C && o < O) v = w[((size_t)o * C + c) * 25 + t];
    wp[idx] = (__bf16)v;
}

// ---------------------------------------------------------------------------
// async global -> LDS (CDNA5 VGLOBAL async encoding, ASYNCcnt).
// lds_base is passed (unreferenced) so the LDS object ESCAPES into the asm:
// without it LLVM proves the shared array is store-free and folds all
// ds_loads from it to undef (observed in round 2).
// ---------------------------------------------------------------------------
__device__ __forceinline__ void async_ld_b64(unsigned lds_off, const void* g,
                                             const void* lds_base) {
    asm volatile("global_load_async_to_lds_b64 %0, %1, off"
                 :: "v"(lds_off), "v"((unsigned long long)g),
                    "v"((unsigned long long)lds_base)
                 : "memory");
}
__device__ __forceinline__ void wait_async0() {
    asm volatile("s_wait_asynccnt 0x0" ::: "memory");
}

// ---------------------------------------------------------------------------
// Implicit-GEMM conv5x5 pad2 + BN + ReLU, NHWC.
//   block = 128 (4 waves); tile = 16 pixels x 64 out-channels
//   grid  = (H*W/16, ceil(O/64), B)
// A tile (16 px x 32 chan) double-buffered in LDS via async DMA; each lane's
// A fragment = two ds_load_b128, B fragment = one contiguous 32B global read.
// ---------------------------------------------------------------------------
#define ASTR 40   // LDS row stride in bf16 (8B-aligned, bank-staggered)

__global__ void __launch_bounds__(128)
k_conv5x5_wmma(const __bf16* __restrict__ in, int Cin, int H, int W,
               const __bf16* __restrict__ wp, int Opad,
               const float* __restrict__ bconv, const float* __restrict__ g,
               const float* __restrict__ e,
               __bf16* __restrict__ out, int Ochoff, int Ot, int O,
               const __bf16* __restrict__ zb) {
    // NOTE: single __shared__ object -> LDS offset 0 (module LDS lowering)
    __shared__ __align__(16) __bf16 At[2][16 * ASTR];

    const int HW    = H * W;
    const int b     = blockIdx.z;
    const int tid   = threadIdx.x;
    const int lane  = tid & 31;
    const int wid   = tid >> 5;
    const int obase = blockIdx.y * 64 + wid * 16;
    const int nn    = lane & 15;           // A:M row / B:N col / D:N col
    const int hh    = lane >> 4;           // half-wave selector
    const int CB    = Cin >> 5;            // 32-channel blocks
    const int KT    = 25 * CB;
    const int p0    = blockIdx.x * 16;

    // producer thread mapping: pixel m = tid/8, channels cc..cc+3
    const int m  = tid >> 3;
    const int cc = (tid & 7) * 4;
    const int p  = p0 + m;
    const int y  = p / W;
    const int x  = p - y * W;

    const __bf16* inb = in + (size_t)b * HW * Cin;

    // visible zero-init of both buffers (real ds_store_b64: keeps the LDS
    // object alive through GlobalOpt; semantically a no-op, DMA overwrites)
    *(unsigned long long*)&At[0][m * ASTR + cc] = 0ull;
    *(unsigned long long*)&At[1][m * ASTR + cc] = 0ull;
    __syncthreads();

    // LDS byte destinations for this thread's 8-byte async store (buf 0/1)
    const unsigned l0 = (unsigned)(m * ASTR + cc) * 2u;
    const unsigned l1 = l0 + (unsigned)(16 * ASTR) * 2u;

    // tap address generator (invalid taps -> zero page; EXEC stays all-1s)
    int stepB = 0;     // bytes to advance per 32-channel block for this tap
    auto tap_ptr = [&](int t) -> const __bf16* {
        int dy = t / 5 - 2;
        int dx = (t - (t / 5) * 5) - 2;
        int yy = y + dy, xx = x + dx;
        bool v = (yy >= 0) & (yy < H) & (xx >= 0) & (xx < W);
        stepB = v ? 64 : 0;
        return v ? (inb + ((size_t)yy * W + xx) * Cin + cc) : zb;
    };

    int tp = 0, cbp = 0;
    const __bf16* gcur = tap_ptr(0);
    async_ld_b64(l0, gcur, At);                   // preload K-step 0

    const __bf16* wptr = wp + ((size_t)(obase + nn < Opad ? obase + nn : 0) << 5)
                            + hh * 16;
    const size_t wstep = (size_t)Opad * 32;

    v8f acc = {};

    for (int k = 0; k < KT; ++k) {
        wait_async0();          // our step-k DMA done
        __syncthreads();        // everyone's step-k tile in LDS; step-(k-1)
                                // fragment reads all retired -> safe to refill
        if (k + 1 < KT) {
            if (++cbp == CB) { cbp = 0; gcur = tap_ptr(++tp); }
            else { gcur = (const __bf16*)((const char*)gcur + stepB); }
            async_ld_b64((k & 1) ? l0 : l1, gcur, At);  // fill other buffer
        }

        // A fragment (contiguous K layout: lane=M, half selects K 0-15/16-31)
        const __bf16* Ab = &At[k & 1][0];
        v8bf alo = *(const v8bf*)&Ab[nn * ASTR + hh * 16];
        v8bf ahi = *(const v8bf*)&Ab[nn * ASTR + hh * 16 + 8];
        v16bf a;
        #pragma unroll
        for (int i = 0; i < 8; ++i) { a[i] = alo[i]; a[8 + i] = ahi[i]; }

        // B fragment: one contiguous 32B read from packed weights
        v16bf bf = *(const v16bf*)wptr;
        __builtin_prefetch(wptr + wstep, 0, 1);
        wptr += wstep;

        acc = __builtin_amdgcn_wmma_f32_16x16x32_bf16(
                  false, a, false, bf, (short)0, acc, false, false);
    }

    // fused epilogue: (acc + bias) * (g*BN_INV) + e, ReLU, NHWC bf16 store
    int o = obase + nn;
    if (o < O) {
        float sc = g[o] * BN_INV;
        float sh = e[o];
        float bb = bconv[o];
        __bf16* ob = out + ((size_t)b * HW + p0) * Ot + Ochoff + o;
        #pragma unroll
        for (int r = 0; r < 8; ++r) {
            int mm = r + hh * 8;               // D layout: M = vgpr + 8*(lane/16)
            float v = (acc[r] + bb) * sc + sh;
            v = v > 0.0f ? v : 0.0f;
            ob[(size_t)mm * Ot] = (__bf16)v;
        }
    }
}

// ---------------------------------------------------------------------------
// 2x2 average pool, NHWC (reads channels [0,C) of a Ctot-channel buffer)
// ---------------------------------------------------------------------------
__global__ void k_avgpool2(const __bf16* __restrict__ in, int Ctot,
                           int C, int H, int W, __bf16* __restrict__ out, int n) {
    int idx = blockIdx.x * blockDim.x + threadIdx.x;
    if (idx >= n) return;
    int ow = W >> 1, oh = H >> 1;
    int c = idx % C;
    int t = idx / C;
    int x = t % ow;
    int y = (t / ow) % oh;
    int b = t / (ow * oh);
    const __bf16* p = in + (((size_t)b * H + 2 * y) * W + 2 * x) * Ctot + c;
    size_t rs = (size_t)W * Ctot;
    float s = (float)p[0] + (float)p[Ctot] + (float)p[rs] + (float)p[rs + Ctot];
    out[idx] = (__bf16)(0.25f * s);
}

// ---------------------------------------------------------------------------
// Bilinear upsample (align_corners=True), NHWC, writes channel sub-range
// ---------------------------------------------------------------------------
__global__ void k_upsample(const __bf16* __restrict__ in, int C, int H, int W,
                           __bf16* __restrict__ out, int choff, int Ctot,
                           int OH, int OW, int n) {
    int idx = blockIdx.x * blockDim.x + threadIdx.x;
    if (idx >= n) return;
    int c = idx % C;
    int t = idx / C;
    int x = t % OW;
    int y = (t / OW) % OH;
    int b = t / (OW * OH);
    float sy = (float)y * (float)(H - 1) / (float)(OH - 1);
    float sx = (float)x * (float)(W - 1) / (float)(OW - 1);
    int y0 = (int)sy; int y1 = y0 + 1 < H ? y0 + 1 : H - 1;
    int x0 = (int)sx; int x1 = x0 + 1 < W ? x0 + 1 : W - 1;
    float wy = sy - (float)y0, wx = sx - (float)x0;
    const __bf16* pb = in + (size_t)b * H * W * C + c;
    float v00 = (float)pb[((size_t)y0 * W + x0) * C];
    float v01 = (float)pb[((size_t)y0 * W + x1) * C];
    float v10 = (float)pb[((size_t)y1 * W + x0) * C];
    float v11 = (float)pb[((size_t)y1 * W + x1) * C];
    float v = (v00 * (1.f - wx) + v01 * wx) * (1.f - wy)
            + (v10 * (1.f - wx) + v11 * wx) * wy;
    out[(((size_t)b * OH + y) * OW + x) * Ctot + choff + c] = (__bf16)v;
}

// ---------------------------------------------------------------------------
// Final: core = 1x1conv(up(c8 -> 512x512), wo) + bo ; out = data * core
// c8 is NHWC (2,256,256,3); output NCHW fp32 to match reference.
// ---------------------------------------------------------------------------
__global__ void k_final(const float* __restrict__ data, const __bf16* __restrict__ c8,
                        const float* __restrict__ wo, const float* __restrict__ bo,
                        float* __restrict__ out, int n) {
    int idx = blockIdx.x * blockDim.x + threadIdx.x;
    if (idx >= n) return;
    const int W = 512, H = 512, SW = 256, SH = 256;
    int x = idx % W;
    int y = (idx / W) % H;
    int o = (idx / (W * H)) % 3;
    int b = idx / (W * H * 3);
    float sy = (float)y * (float)(SH - 1) / (float)(H - 1);
    float sx = (float)x * (float)(SW - 1) / (float)(W - 1);
    int y0 = (int)sy; int y1 = y0 + 1 < SH ? y0 + 1 : SH - 1;
    int x0 = (int)sx; int x1 = x0 + 1 < SW ? x0 + 1 : SW - 1;
    float wy = sy - (float)y0, wx = sx - (float)x0;
    const __bf16* pb = c8 + (size_t)b * SH * SW * 3;
    float core = bo[o];
    #pragma unroll
    for (int i = 0; i < 3; ++i) {
        float v00 = (float)pb[((size_t)y0 * SW + x0) * 3 + i];
        float v01 = (float)pb[((size_t)y0 * SW + x1) * 3 + i];
        float v10 = (float)pb[((size_t)y1 * SW + x0) * 3 + i];
        float v11 = (float)pb[((size_t)y1 * SW + x1) * 3 + i];
        float s = (v00 * (1.f - wx) + v01 * wx) * (1.f - wy)
                + (v10 * (1.f - wx) + v11 * wx) * wy;
        core += wo[o * 3 + i] * s;
    }
    out[idx] = data[idx] * core;
}

// ---------------------------------------------------------------------------
// Host-side orchestration
// ---------------------------------------------------------------------------
extern "C" void kernel_launch(void* const* d_in, const int* in_sizes, int n_in,
                              void* d_out, int out_size, void* d_ws, size_t ws_size,
                              hipStream_t stream) {
    (void)in_sizes; (void)n_in; (void)out_size; (void)ws_size;

    const float* data = (const float*)d_in[0];
    const float *Wt[8], *Bc[8], *Gm[8], *Ep[8];
    for (int l = 0; l < 8; ++l) {
        Wt[l] = (const float*)d_in[1 + l * 4];
        Bc[l] = (const float*)d_in[2 + l * 4];
        Gm[l] = (const float*)d_in[3 + l * 4];
        Ep[l] = (const float*)d_in[4 + l * 4];
    }
    const float* wo = (const float*)d_in[33];
    const float* bo = (const float*)d_in[34];
    float* outp = (float*)d_out;

    // per-layer dims
    const int Ca[8]   = {3, 64, 128, 256, 512, 1024, 768, 384};   // actual Cin
    const int Cr[8]   = {32, 64, 128, 256, 512, 1024, 768, 384};  // rounded Cin
    const int Oc[8]   = {64, 128, 256, 512, 512, 512, 256, 3};
    const int Hh[8]   = {512, 256, 128, 64, 32, 64, 128, 256};
    const int Opad[8] = {64, 128, 256, 512, 512, 512, 256, 16};

    size_t off = 0;
    auto alloc = [&](size_t bytes) -> char* {
        size_t o = (off + 255) & ~(size_t)255;
        off = o + bytes;
        return (char*)d_ws + o;
    };
    __bf16* zb   = (__bf16*)alloc(256);                            // zero page
    __bf16* xin  = (__bf16*)alloc((size_t)2 * 512 * 512 * 32 * 2); // NHWC padded
    __bf16* wp[8];
    for (int l = 0; l < 8; ++l)
        wp[l] = (__bf16*)alloc((size_t)25 * Cr[l] * Opad[l] * 2);
    __bf16* c1   = (__bf16*)alloc((size_t)2 * 512 * 512 * 64  * 2);
    __bf16* p1   = (__bf16*)alloc((size_t)2 * 256 * 256 * 64  * 2);
    __bf16* cat8 = (__bf16*)alloc((size_t)2 * 256 * 256 * 384 * 2);
    __bf16* p2   = (__bf16*)alloc((size_t)2 * 128 * 128 * 128 * 2);
    __bf16* cat7 = (__bf16*)alloc((size_t)2 * 128 * 128 * 768 * 2);
    __bf16* p3   = (__bf16*)alloc((size_t)2 * 64  * 64  * 256 * 2);
    __bf16* cat6 = (__bf16*)alloc((size_t)2 * 64  * 64  * 1024* 2);
    __bf16* p4   = (__bf16*)alloc((size_t)2 * 32  * 32  * 512 * 2);
    __bf16* c5   = (__bf16*)alloc((size_t)2 * 32  * 32  * 512 * 2);
    __bf16* c6   = (__bf16*)alloc((size_t)2 * 64  * 64  * 512 * 2);
    __bf16* c7   = (__bf16*)alloc((size_t)2 * 128 * 128 * 256 * 2);
    __bf16* c8   = (__bf16*)alloc((size_t)2 * 256 * 256 * 3   * 2);

    // zero page + input convert/pad
    k_zero_bf16<<<1, 128, 0, stream>>>(zb, 128);
    {
        int n = 2 * 512 * 512 * 32;
        k_cvt_pad<<<(n + 255) / 256, 256, 0, stream>>>(data, xin, n);
    }
    // pack weights
    for (int l = 0; l < 8; ++l) {
        int total = 25 * Cr[l] * Opad[l];
        k_pack_w<<<(total + 255) / 256, 256, 0, stream>>>(
            Wt[l], wp[l], Ca[l], Cr[l], Oc[l], Opad[l], total);
    }

    auto conv = [&](int l, const __bf16* in, __bf16* out, int choff, int Ot) {
        int H = Hh[l];
        dim3 grid(H * H / 16, (Oc[l] + 63) / 64, 2);
        k_conv5x5_wmma<<<grid, 128, 0, stream>>>(
            in, Cr[l], H, H, wp[l], Opad[l],
            Bc[l], Gm[l], Ep[l], out, choff, Ot, Oc[l], zb);
    };
    auto pool = [&](const __bf16* in, int Ctot, int C, int H, __bf16* out) {
        int n = 2 * C * (H / 2) * (H / 2);
        k_avgpool2<<<(n + 255) / 256, 256, 0, stream>>>(in, Ctot, C, H, H, out, n);
    };
    auto up = [&](const __bf16* in, int C, int H, __bf16* out, int choff,
                  int Ctot, int OH) {
        int n = 2 * C * OH * OH;
        k_upsample<<<(n + 255) / 256, 256, 0, stream>>>(in, C, H, H, out, choff,
                                                        Ctot, OH, OH, n);
    };

    // encoder
    conv(0, xin, c1, 0, 64);                 // c1 (2,512,512,64)
    pool(c1, 64, 64, 512, p1);               // p1
    conv(1, p1, cat8, 0, 384);               // c2 -> cat8[...,0:128)
    pool(cat8, 384, 128, 256, p2);           // p2
    conv(2, p2, cat7, 0, 768);               // c3 -> cat7[...,0:256)
    pool(cat7, 768, 256, 128, p3);           // p3
    conv(3, p3, cat6, 0, 1024);              // c4 -> cat6[...,0:512)
    pool(cat6, 1024, 512, 64, p4);           // p4
    conv(4, p4, c5, 0, 512);                 // c5 (2,32,32,512)

    // decoder
    up(c5, 512, 32, cat6, 512, 1024, 64);    // cat6[...,512:1024)
    conv(5, cat6, c6, 0, 512);               // c6
    up(c6, 512, 64, cat7, 256, 768, 128);    // cat7[...,256:768)
    conv(6, cat7, c7, 0, 256);               // c7
    up(c7, 256, 128, cat8, 128, 384, 256);   // cat8[...,128:384)
    conv(7, cat8, c8, 0, 3);                 // c8 (2,256,256,3)

    // final fused upsample + 1x1 conv + multiply
    {
        int n = 2 * 3 * 512 * 512;
        k_final<<<(n + 255) / 256, 256, 0, stream>>>(data, c8, wo, bo, outp, n);
    }
}
